// BiAttn_28484223107487
// MI455X (gfx1250) — matrix operand order repaired and checked by reading, not compiled, not run on hardware
//
#include <hip/hip_runtime.h>

#define BB 32
#define L1 2048
#define L2 512
#define DD 256

typedef float v8f  __attribute__((ext_vector_type(8)));
typedef float v4f  __attribute__((ext_vector_type(4)));
typedef __bf16 v16bf __attribute__((ext_vector_type(16)));
typedef __bf16 v8bf  __attribute__((ext_vector_type(8)));

#define XW_OFF   0
#define YR_OFF   67584
#define YT_OFF   135168
#define SVEC_OFF 204800
#define SX_OFF   206848
#define SMEM_BYTES 207360
#define XW_STRIDE 264   /* bf16 elems per row (528B, 4-bank shift/row) */
#define YR_STRIDE 264
#define YT_STRIDE 136   /* 272B rows for transposed y chunk */
#define OB_STRIDE 260   /* f32 elems per row of output staging */
#define NEGV    (-1e30f)
#define MASKNEG (-3.0e38f)

static __device__ __forceinline__ v8f wmma_bf16(v16bf a, v16bf b, v8f c) {
  return __builtin_amdgcn_wmma_f32_16x16x32_bf16(false, a, false, b, (short)0, c, false, false);
}
// A-operand per-lane pattern: 8 elems at +0, 8 elems at +16 (elements)
static __device__ __forceinline__ v16bf ld16_split(const __bf16* p) {
  v8bf lo = *(const v8bf*)p;
  v8bf hi = *(const v8bf*)(p + 16);
  return __builtin_shufflevector(lo, hi, 0,1,2,3,4,5,6,7,8,9,10,11,12,13,14,15);
}
// per-lane load of 16 contiguous elems (contiguous k order k = 16h + s; Y^T A operand here)
static __device__ __forceinline__ v16bf ld16_contig(const __bf16* p) {
  v8bf lo = *(const v8bf*)p;
  v8bf hi = *(const v8bf*)(p + 8);
  return __builtin_shufflevector(lo, hi, 0,1,2,3,4,5,6,7,8,9,10,11,12,13,14,15);
}

__global__ void __launch_bounds__(256)
biattn_kernel(const float* __restrict__ x, const float* __restrict__ y,
              const unsigned char* __restrict__ xmask, const unsigned char* __restrict__ ymask,
              const float* __restrict__ w_x, const float* __restrict__ w_y,
              const float* __restrict__ w_dot, const float* __restrict__ bias,
              float* __restrict__ out_c2q, float* __restrict__ out_xc2q,
              float* __restrict__ mi_ws)
{
  extern __shared__ char smem[];
  __bf16* XW = (__bf16*)(smem + XW_OFF);    // 128 x-rows * w_dot, bf16
  __bf16* YR = (__bf16*)(smem + YR_OFF);    // y chunk row-major (128 x 256)
  __bf16* YT = (__bf16*)(smem + YT_OFF);    // y chunk transposed (256 x 128)
  float*  SVEC = (float*)(smem + SVEC_OFF); // sy + y-mask, 512
  float*  SX   = (float*)(smem + SX_OFF);   // x@w_x for block rows, 128

  const int tid  = threadIdx.x;
  const int wave = tid >> 5, lane = tid & 31;
  const int g    = lane >> 4, lm = lane & 15;
  const int b    = blockIdx.y;
  const int row0 = blockIdx.x * 128;

  // ---------------- prologue: SVEC, SX, XW ----------------
  for (int rr = 0; rr < 2; ++rr) {
    int l2 = tid + rr * 256;
    const float4* yr = (const float4*)(y + ((size_t)(b * L2 + l2)) * DD);
    const float4* wy = (const float4*)w_y;
    float acc = 0.f;
    for (int k2 = 0; k2 < 64; ++k2) {
      float4 a = yr[k2], w = wy[k2];
      acc += a.x*w.x + a.y*w.y + a.z*w.z + a.w*w.w;
    }
    SVEC[l2] = ymask[b * L2 + l2] ? MASKNEG : acc;
  }
  if (tid < 128) {
    const float4* xr = (const float4*)(x + ((size_t)(b * L1 + row0 + tid)) * DD);
    const float4* wx = (const float4*)w_x;
    float acc = 0.f;
    for (int k2 = 0; k2 < 64; ++k2) {
      float4 a = xr[k2], w = wx[k2];
      acc += a.x*w.x + a.y*w.y + a.z*w.z + a.w*w.w;
    }
    SX[tid] = acc;
  }
  {
    int dp = tid & 127;
    float2 wd = *(const float2*)(w_dot + 2 * dp);
    for (int it = 0; it < 64; ++it) {
      int r = it * 2 + (tid >> 7);
      float2 v = *(const float2*)(x + ((size_t)(b * L1 + row0 + r)) * DD + 2 * dp);
      __bf16* p = XW + r * XW_STRIDE + 2 * dp;
      p[0] = (__bf16)(v.x * wd.x);
      p[1] = (__bf16)(v.y * wd.y);
    }
  }

  const int myrow = row0 + wave * 16 + lm;
  const unsigned char xm = xmask[b * L1 + myrow];
  const __bf16* xwrow = XW + (wave * 16 + lm) * XW_STRIDE;

  // ---------------- sweep 1: row max of (sdot + sy, masked) ----------------
  float maxacc = NEGV;
  for (int ch = 0; ch < 4; ++ch) {
    __syncthreads();
    {
      int dp = tid & 127;
      for (int it = 0; it < 64; ++it) {
        int r = it * 2 + (tid >> 7);
        float2 v = *(const float2*)(y + ((size_t)(b * L2 + ch * 128 + r)) * DD + 2 * dp);
        __bf16* p = YR + r * YR_STRIDE + 2 * dp;
        p[0] = (__bf16)v.x; p[1] = (__bf16)v.y;
      }
    }
    __syncthreads();
    for (int jt = 0; jt < 8; ++jt) {  // S^T tile: rows l2 = ch*128+16*jt.., cols m
      v8f acc = {};
      const __bf16* arow = YR + (16 * jt + lm) * YR_STRIDE + 8 * g;
      const __bf16* brow = xwrow + 8 * g;
      #pragma unroll
      for (int kb = 0; kb < 8; ++kb) {
        v16bf af = ld16_split(arow + 32 * kb);
        v16bf bf = ld16_split(brow + 32 * kb);
        acc = wmma_bf16(af, bf, acc);
      }
      const float* sp = SVEC + ch * 128 + 16 * jt + 8 * g;
      v4f s0 = *(const v4f*)sp, s1 = *(const v4f*)(sp + 4);
      #pragma unroll
      for (int j = 0; j < 8; ++j) {
        float sv = (j < 4) ? s0[j] : s1[j - 4];
        maxacc = fmaxf(maxacc, acc[j] + sv);
      }
    }
  }
  float rowmax = fmaxf(maxacc, __shfl_xor(maxacc, 16, 32));
  if (lane < 16)
    mi_ws[b * L1 + myrow] = xm ? NEGV : (rowmax + SX[wave * 16 + lm] + bias[0]);

  // ---------------- sweep 2: exp + O^T accumulate ----------------
  v8f oacc[16];
  #pragma unroll
  for (int t = 0; t < 16; ++t) oacc[t] = (v8f){};
  float sumacc = 0.f;

  for (int ch = 0; ch < 4; ++ch) {
    __syncthreads();
    {
      int dp = tid & 127;
      for (int it = 0; it < 64; ++it) {
        int r = it * 2 + (tid >> 7);
        float2 v = *(const float2*)(y + ((size_t)(b * L2 + ch * 128 + r)) * DD + 2 * dp);
        __bf16 b0 = (__bf16)v.x, b1 = (__bf16)v.y;
        __bf16* p = YR + r * YR_STRIDE + 2 * dp;
        p[0] = b0; p[1] = b1;
        YT[(2 * dp) * YT_STRIDE + r]     = b0;
        YT[(2 * dp + 1) * YT_STRIDE + r] = b1;
      }
    }
    __syncthreads();
    #pragma unroll 1
    for (int c = 0; c < 4; ++c) {      // 32 l2-values per step (tiles 2c, 2c+1)
      v8f p01[2];
      #pragma unroll
      for (int h = 0; h < 2; ++h) {
        int jt = 2 * c + h;
        v8f acc = {};
        const __bf16* arow = YR + (16 * jt + lm) * YR_STRIDE + 8 * g;
        const __bf16* brow = xwrow + 8 * g;
        #pragma unroll
        for (int kb = 0; kb < 8; ++kb) {
          v16bf af = ld16_split(arow + 32 * kb);
          v16bf bf = ld16_split(brow + 32 * kb);
          acc = wmma_bf16(af, bf, acc);
        }
        const float* sp = SVEC + ch * 128 + 16 * jt + 8 * g;
        v4f s0 = *(const v4f*)sp, s1 = *(const v4f*)(sp + 4);
        #pragma unroll
        for (int j = 0; j < 8; ++j) {
          float sv = (j < 4) ? s0[j] : s1[j - 4];
          float e = __expf(acc[j] + sv - rowmax);
          sumacc += e;
          acc[j] = e;
        }
        p01[h] = acc;
      }
      // in-register C-layout -> B-layout repack (pair lanes L <-> L^16)
      v8f own, snd, rcv;
      #pragma unroll
      for (int j = 0; j < 8; ++j) {
        own[j] = g ? p01[1][j] : p01[0][j];
        snd[j] = g ? p01[0][j] : p01[1][j];
      }
      #pragma unroll
      for (int j = 0; j < 8; ++j) rcv[j] = __shfl_xor(snd[j], 16, 32);
      v16bf pf;
      #pragma unroll
      for (int j = 0; j < 8; ++j) {
        float lo = g ? rcv[j] : own[j];
        float hi = g ? own[j] : rcv[j];
        pf[j]     = (__bf16)lo;
        pf[j + 8] = (__bf16)hi;
      }
      // O^T[d, m] += Y^T tile x P
      #pragma unroll
      for (int t = 0; t < 16; ++t) {
        const __bf16* ap = YT + (16 * t + lm) * YT_STRIDE + 32 * c + 16 * g;
        v16bf af = ld16_contig(ap);
        oacc[t] = wmma_bf16(af, pf, oacc[t]);
      }
    }
  }
  float rowsum = sumacc + __shfl_xor(sumacc, 16, 32);
  float scale = (xm || !(rowsum > 1e-37f)) ? 0.f : 1.f / rowsum;

  // ---------------- epilogue: transpose O^T via LDS, coalesced stores ----------------
  __syncthreads();
  float* ob = (float*)(smem + YR_OFF) + wave * (16 * OB_STRIDE);
  #pragma unroll
  for (int t = 0; t < 16; ++t)
    #pragma unroll
    for (int j = 0; j < 8; ++j)
      ob[lm * OB_STRIDE + 16 * t + j + 8 * g] = oacc[t][j] * scale;
  __syncthreads();
  for (int it = 0; it < 32; ++it) {
    int idx = it * 32 + lane;
    int m = idx >> 6;
    int d4 = (idx & 63) << 2;
    float4 v = *(const float4*)(ob + m * OB_STRIDE + d4);
    size_t o = ((size_t)(b * L1 + row0 + wave * 16 + m)) * DD + d4;
    float4 xv = *(const float4*)(x + o);
    *(float4*)(out_c2q + o) = v;
    float4 x2; x2.x = v.x * xv.x; x2.y = v.y * xv.y; x2.z = v.z * xv.z; x2.w = v.w * xv.w;
    *(float4*)(out_xc2q + o) = x2;
  }
}

// ---------------- kernel 2: beta = softmax_L1(m_i), q2c = beta @ x ----------------
__global__ void __launch_bounds__(256)
q2c_kernel(const float* __restrict__ x, const float* __restrict__ mi,
           float* __restrict__ out_q2c)
{
  __shared__ float beta[L1];
  __shared__ float red[256];
  int b = blockIdx.x, tid = threadIdx.x;
  const float* m = mi + b * L1;
  float mx = -3.4e38f;
  for (int k = 0; k < 8; ++k) mx = fmaxf(mx, m[tid + 256 * k]);
  red[tid] = mx; __syncthreads();
  for (int s = 128; s > 0; s >>= 1) {
    if (tid < s) red[tid] = fmaxf(red[tid], red[tid + s]);
    __syncthreads();
  }
  float M = red[0]; __syncthreads();
  float sm = 0.f;
  for (int k = 0; k < 8; ++k) {
    float e = __expf(m[tid + 256 * k] - M);
    beta[tid + 256 * k] = e;
    sm += e;
  }
  red[tid] = sm; __syncthreads();
  for (int s = 128; s > 0; s >>= 1) {
    if (tid < s) red[tid] += red[tid + s];
    __syncthreads();
  }
  float inv = 1.f / red[0];
  float acc = 0.f;
  const float* xb = x + (size_t)b * L1 * DD + tid;
  for (int i = 0; i < L1; ++i) acc += beta[i] * xb[(size_t)i * DD];
  out_q2c[b * DD + tid] = acc * inv;
}

extern "C" void kernel_launch(void* const* d_in, const int* in_sizes, int n_in,
                              void* d_out, int out_size, void* d_ws, size_t ws_size,
                              hipStream_t stream) {
  (void)in_sizes; (void)n_in; (void)out_size; (void)ws_size;
  const float* x = (const float*)d_in[0];
  const float* y = (const float*)d_in[1];
  const unsigned char* xmask = (const unsigned char*)d_in[2];
  const unsigned char* ymask = (const unsigned char*)d_in[3];
  const float* w_x   = (const float*)d_in[4];
  const float* w_y   = (const float*)d_in[5];
  const float* w_dot = (const float*)d_in[6];
  const float* bias  = (const float*)d_in[7];

  float* out      = (float*)d_out;
  float* out_c2q  = out;
  float* out_q2c  = out + (size_t)BB * L1 * DD;
  float* out_xc2q = out_q2c + BB * DD;
  float* mi_ws    = (float*)d_ws;   // B*L1 f32 = 256KB

  hipFuncSetAttribute((const void*)biattn_kernel,
                      hipFuncAttributeMaxDynamicSharedMemorySize, SMEM_BYTES);

  dim3 grid(L1 / 128, BB);
  biattn_kernel<<<grid, 256, SMEM_BYTES, stream>>>(
      x, y, xmask, ymask, w_x, w_y, w_dot, bias, out_c2q, out_xc2q, mi_ws);
  q2c_kernel<<<BB, 256, 0, stream>>>(x, mi_ws, out_q2c);
}
